// MsaHmmLayer_88330297409626
// MI455X (gfx1250) — compile-verified, matrix operand order checked by reading, not run on hardware
//
#include <hip/hip_runtime.h>
#include <hip/hip_bf16.h>

typedef _Float16 h16v __attribute__((ext_vector_type(16)));
typedef _Float16 h8v  __attribute__((ext_vector_type(8)));
typedef _Float16 h4v  __attribute__((ext_vector_type(4)));
typedef float    f8v  __attribute__((ext_vector_type(8)));
typedef float    f4v  __attribute__((ext_vector_type(4)));

#define NM   2
#define BT   32
#define BR   16   // batch rows per block (one M tile)
#define LEN  512
#define NS   26
#define Q    512
#define NTL  32   // N tiles (512/16)
#define KTL  16   // K tiles (512/32)
#define KC   4    // k-tiles cached in registers
#define POST_ELEMS (2ull*32*512*512)
#define RSTRIDE ((size_t)LEN * Q * 2)   // shorts between consecutive batch rows

// ---------- fragment helpers (16-bit WMMA per-lane layouts, ISA 7.12.2) ----------

__device__ __forceinline__ h16v h16cat(h8v lo, h8v hi) {
  h16v r;
#pragma unroll
  for (int i = 0; i < 8; ++i) { r[i] = lo[i]; r[i + 8] = hi[i]; }
  return r;
}

__device__ __forceinline__ h16v ldfrag(const _Float16* base, int lane) {
  const h8v* p = (const h8v*)(base + lane * 16);
  return h16cat(p[0], p[1]);
}

__device__ __forceinline__ h16v ldrowfrag(const _Float16* V, int kt, int lane) {
  int row = lane & 15;
  const _Float16* p = V + row * Q + kt * 32 + ((lane >> 4) & 1) * 8;
  const h8v* q = (const h8v*)p;
  return h16cat(q[0], q[2]);
}

__device__ __forceinline__ f8v wmma_f16(h16v a, h16v b, f8v c) {
  return __builtin_amdgcn_wmma_f32_16x16x32_f16(false, a, false, b, (short)0, c,
                                                false, false);
}

__device__ __forceinline__ int frag_pos(int ki, int nj) {
  int lane = (nj & 15) + 16 * ((ki >> 3) & 1);
  int slot = (ki & 7) | ((ki >> 4) << 3);
  return lane * 16 + slot;
}

// float -> bf16 bits, round-half-up (positive finite inputs): 1 VALU op + store
__device__ __forceinline__ unsigned short f2bf(float x) {
  return (unsigned short)((__float_as_uint(x) + 0x8000u) >> 16);
}

// ---------- ds_swizzle XOR-mode wave reductions (ISA 11.1/11.2) ----------

template <int IMM>
__device__ __forceinline__ float swzmax(float x) {
  return fmaxf(x, __int_as_float(__builtin_amdgcn_ds_swizzle(__float_as_int(x), IMM)));
}
template <int IMM>
__device__ __forceinline__ float swzadd(float x) {
  return x + __int_as_float(__builtin_amdgcn_ds_swizzle(__float_as_int(x), IMM));
}

// Fused reduce+scale: wave w owns row w. Row max ends up in all lanes,
// the same wave rescales the values it already holds. Returns row max.
__device__ __forceinline__ float fuse_reduce_scale(const float* Pb, _Float16* Vh,
                                                   int tid) {
  int b = tid >> 5, sg = tid & 31;
  const f4v* p4 = (const f4v*)(Pb + b * Q + sg * 16);
  f4v v[4];
#pragma unroll
  for (int k = 0; k < 4; ++k) v[k] = p4[k];
  float mx = 0.f;
#pragma unroll
  for (int k = 0; k < 4; ++k)
    mx = fmaxf(fmaxf(fmaxf(mx, v[k][0]), fmaxf(v[k][1], v[k][2])), v[k][3]);
  mx = swzmax<0x401F>(mx);   // SWAPX16
  mx = swzmax<0x201F>(mx);   // SWAPX8
  mx = swzmax<0x101F>(mx);   // SWAPX4
  mx = swzmax<0x081F>(mx);   // SWAPX2
  mx = swzmax<0x041F>(mx);   // SWAPX1
  mx = fmaxf(mx, 1e-37f);
  float inv = 1.0f / mx;
  h4v* q4 = (h4v*)(Vh + b * Q + sg * 16);
#pragma unroll
  for (int k = 0; k < 4; ++k) {
    h4v o;
#pragma unroll
    for (int i = 0; i < 4; ++i) o[i] = (_Float16)(v[k][i] * inv);
    q4[k] = o;
  }
  return mx;
}

// ---------- preprocessing kernels ----------

__global__ void __launch_bounds__(512)
prep_trans(const float* __restrict__ tl, _Float16* __restrict__ Afwd,
           _Float16* __restrict__ Abwd) {
  int m = blockIdx.x;
  int i = threadIdx.x;
  const float* row = tl + ((size_t)m * Q + i) * Q;
  float mx = -3.402823e38f;
  for (int j = 0; j < Q; ++j) mx = fmaxf(mx, row[j]);
  float s = 0.f;
  for (int j = 0; j < Q; ++j) s += __expf(row[j] - mx);
  float inv = 1.0f / s;
  _Float16* Af = Afwd + (size_t)m * NTL * KTL * 512;
  _Float16* Ab = Abwd + (size_t)m * NTL * KTL * 512;
  int kt = i >> 5, ki = i & 31, nt2 = i >> 4;
  for (int j = 0; j < Q; ++j) {
    _Float16 p = (_Float16)(__expf(row[j] - mx) * inv);
    int ntf = j >> 4;
    Af[(size_t)(ntf * KTL + kt) * 512 + frag_pos(ki, j & 15)] = p;
    int kt2 = j >> 5, kj = j & 31;
    Ab[(size_t)(nt2 * KTL + kt2) * 512 + frag_pos(kj, i & 15)] = p;
  }
}

__global__ void __launch_bounds__(512)
prep_emis(const float* __restrict__ el, _Float16* __restrict__ BmFrag) {
  int m = blockIdx.x;
  int j = threadIdx.x;
  const float* row = el + ((size_t)m * Q + j) * NS;
  float mx = -3.402823e38f;
  for (int s = 0; s < NS; ++s) mx = fmaxf(mx, row[s]);
  float sum = 0.f;
  for (int s = 0; s < NS; ++s) sum += __expf(row[s] - mx);
  float inv = 1.0f / sum;
  _Float16* F = BmFrag + (size_t)m * NTL * 512 + (size_t)(j >> 4) * 512;
  for (int s = 0; s < 32; ++s) {
    _Float16 p = (s < NS) ? (_Float16)(__expf(row[s] - mx) * inv) : (_Float16)0.f;
    F[frag_pos(s, j & 15)] = p;
  }
}

__global__ void __launch_bounds__(256)
prep_inputs(const float* __restrict__ in, _Float16* __restrict__ inFrag) {
  int idx = blockIdx.x * blockDim.x + threadIdx.x;
  if (idx >= NM * LEN * BT) return;
  int b = idx & 31;
  int t = (idx >> 5) & (LEN - 1);
  int m = idx >> 14;
  const float* src = in + (((size_t)(m * BT + b)) * LEN + t) * NS;
  _Float16* F = inFrag + ((size_t)m * LEN + t) * 1024 + (size_t)(b >> 4) * 512;
  for (int s = 0; s < 32; ++s) {
    _Float16 p = (s < NS) ? (_Float16)src[s] : (_Float16)0.f;
    F[frag_pos(s, b & 15)] = p;
  }
}

__global__ void prep_init(const float* __restrict__ il, float* __restrict__ pInit) {
  int m = threadIdx.x;
  if (m >= NM) return;
  const float* row = il + m * Q;
  float mx = -3.402823e38f;
  for (int j = 0; j < Q; ++j) mx = fmaxf(mx, row[j]);
  float s = 0.f;
  for (int j = 0; j < Q; ++j) s += __expf(row[j] - mx);
  float inv = 1.0f / s;
  for (int j = 0; j < Q; ++j) pInit[m * Q + j] = __expf(row[j] - mx) * inv;
}

// ---------- fused forward+backward chains (8 concurrent blocks) ----------

__global__ void __launch_bounds__(512)
hmm_chains(const _Float16* __restrict__ Afwd,
           const _Float16* __restrict__ Abwd,
           const _Float16* __restrict__ inFrag,
           const _Float16* __restrict__ BmFrag,
           const float* __restrict__ pInit,
           float* __restrict__ dOut,
           float* __restrict__ loglikWs,
           float* __restrict__ rmFg,
           float* __restrict__ rmBg) {
  __shared__ __align__(16) _Float16 Vh[BR * Q];
  __shared__ __align__(16) float Pb[BR * Q];

  const int dir = blockIdx.x >> 2;            // 0 = forward, 1 = backward
  const int m = (blockIdx.x >> 1) & 1;
  const int c = blockIdx.x & 1;
  const int tid = threadIdx.x;
  const int wv = tid >> 5, lane = tid & 31;
  const int l15 = lane & 15, lhi = (lane >> 4) & 1;
  const _Float16* inF = inFrag + (size_t)m * LEN * 1024 + (size_t)c * 512;
  const _Float16* BmF = BmFrag + (size_t)m * NTL * 512;
  const int gb0 = m * BT + c * BR;
  unsigned short* outs = (unsigned short*)dOut + (size_t)gb0 * LEN * Q * 2;
  const int ncol[2] = { wv, wv + 16 };
  const _Float16* Am = (dir == 0 ? Afwd : Abwd) + (size_t)m * NTL * KTL * 512;
  float* rmG = (dir == 0 ? rmFg : rmBg) + (size_t)(gb0 + wv) * LEN;

  // t-invariant fragments cached in registers
  h16v cA0[KC], cA1[KC];
#pragma unroll
  for (int kt = 0; kt < KC; ++kt) {
    cA0[kt] = ldfrag(Am + ((size_t)ncol[0] * KTL + kt) * 512, lane);
    cA1[kt] = ldfrag(Am + ((size_t)ncol[1] * KTL + kt) * 512, lane);
  }
  h16v cBm0 = ldfrag(BmF + (size_t)ncol[0] * 512, lane);
  h16v cBm1 = ldfrag(BmF + (size_t)ncol[1] * 512, lane);

  // per-thread store bases; per-r displacements are compile-time immediates
  float* pb0 = Pb + (8 * lhi) * Q + l15 + ncol[0] * 16;   // + r*Q
  float* pb1 = pb0 + 256;
  unsigned short* gq0 =
      outs + ((size_t)(8 * lhi) * LEN * Q + l15 + (size_t)ncol[0] * 16) * 2;
  unsigned short* gq1 = gq0 + 512;                        // ncol[1] = ncol[0]+16

  float rmacc = 0.f;     // running log-scale of this wave's row (uniform in wave)
  float lastmx = 1.f;

  if (dir == 0) {
    // =============================== FORWARD ===============================
    const float* pi = pInit + m * Q;
    {   // t = 0: P0 = pInit[j] * em
      h16v a = ldfrag(inF, lane);
      f8v z = {0.f, 0.f, 0.f, 0.f, 0.f, 0.f, 0.f, 0.f};
      f8v e0 = wmma_f16(a, cBm0, z);
      f8v e1 = wmma_f16(a, cBm1, z);
      float pj0 = pi[ncol[0] * 16 + l15];
      float pj1 = pi[ncol[1] * 16 + l15];
#pragma unroll
      for (int r = 0; r < 8; ++r) {
        float P0 = pj0 * e0[r], P1 = pj1 * e1[r];
        pb0[r * Q] = P0;
        pb1[r * Q] = P1;
        gq0[(size_t)r * RSTRIDE] = f2bf(P0);
        gq1[(size_t)r * RSTRIDE] = f2bf(P1);
      }
    }
    __syncthreads();
    {
      float mx = fuse_reduce_scale(Pb, Vh, tid);
      if (lane == 0) rmG[0] = 0.f;
      rmacc = __logf(mx);
      lastmx = mx;
    }
    __syncthreads();

    h16v a = ldfrag(inF + 1024, lane);        // emission fragment for t=1
    unsigned short* gp0 = gq0 + Q * 2;
    unsigned short* gp1 = gq1 + Q * 2;
    for (int t = 1; t < LEN; ++t) {
      f8v z = {0.f, 0.f, 0.f, 0.f, 0.f, 0.f, 0.f, 0.f};
      f8v acc0 = z, acc1 = z;
#pragma unroll
      for (int kt = 0; kt < KC; ++kt) {
        h16v v = ldrowfrag(Vh, kt, lane);
        acc0 = wmma_f16(v, cA0[kt], acc0);
        acc1 = wmma_f16(v, cA1[kt], acc1);
      }
#pragma unroll 4
      for (int kt = KC; kt < KTL; ++kt) {
        h16v v  = ldrowfrag(Vh, kt, lane);
        h16v b0 = ldfrag(Am + ((size_t)ncol[0] * KTL + kt) * 512, lane);
        h16v b1 = ldfrag(Am + ((size_t)ncol[1] * KTL + kt) * 512, lane);
        acc0 = wmma_f16(v, b0, acc0);
        acc1 = wmma_f16(v, b1, acc1);
      }
      f8v em0 = wmma_f16(a, cBm0, z);
      f8v em1 = wmma_f16(a, cBm1, z);
#pragma unroll
      for (int r = 0; r < 8; ++r) {
        float P0 = acc0[r] * em0[r], P1 = acc1[r] * em1[r];
        pb0[r * Q] = P0;
        pb1[r * Q] = P1;
        gp0[(size_t)r * RSTRIDE] = f2bf(P0);
        gp1[(size_t)r * RSTRIDE] = f2bf(P1);
      }
      int tn = (t + 1 < LEN) ? t + 1 : t;
      a = ldfrag(inF + (size_t)tn * 1024, lane);   // hidden behind barrier
      gp0 += Q * 2;
      gp1 += Q * 2;
      __syncthreads();
      {
        float mx = fuse_reduce_scale(Pb, Vh, tid);
        if (lane == 0) rmG[t] = rmacc;
        rmacc += __logf(mx);
        lastmx = mx;
      }
      __syncthreads();
    }

    {   // loglik = rmacc + log(sum(P) / lastmx)
      int b = tid >> 5, sg = tid & 31;
      const f4v* p4 = (const f4v*)(Pb + b * Q + sg * 16);
      float s = 0.f;
#pragma unroll
      for (int k = 0; k < 4; ++k) {
        f4v v = p4[k];
        s += (v[0] + v[1]) + (v[2] + v[3]);
      }
      s = swzadd<0x401F>(s); s = swzadd<0x201F>(s); s = swzadd<0x101F>(s);
      s = swzadd<0x081F>(s); s = swzadd<0x041F>(s);
      if (sg == 0) {
        float ll = rmacc + __logf(s / lastmx);
        int gb = gb0 + b;
        loglikWs[gb] = ll;
        dOut[POST_ELEMS + gb] = ll;
      }
    }
  } else {
    // =============================== BACKWARD ==============================
    {   // t = L-1: beta = 0 ; P = em ; Sb == 1
      h16v a = ldfrag(inF + (size_t)(LEN - 1) * 1024, lane);
      f8v z = {0.f, 0.f, 0.f, 0.f, 0.f, 0.f, 0.f, 0.f};
      f8v e0 = wmma_f16(a, cBm0, z);
      f8v e1 = wmma_f16(a, cBm1, z);
      unsigned short* gt0 = gq0 + (size_t)(LEN - 1) * Q * 2;
      unsigned short* gt1 = gq1 + (size_t)(LEN - 1) * Q * 2;
#pragma unroll
      for (int r = 0; r < 8; ++r) {
        pb0[r * Q] = e0[r];
        pb1[r * Q] = e1[r];
        gt0[(size_t)r * RSTRIDE + 1] = 0x3F80;   // bf16(1.0)
        gt1[(size_t)r * RSTRIDE + 1] = 0x3F80;
      }
    }
    __syncthreads();
    {
      float mx = fuse_reduce_scale(Pb, Vh, tid);
      if (lane == 0) rmG[LEN - 1] = 0.f;
      rmacc = __logf(mx);
    }
    __syncthreads();

    h16v a = ldfrag(inF + (size_t)(LEN - 2) * 1024, lane);
    unsigned short* gp0 = gq0 + (size_t)(LEN - 2) * Q * 2;
    unsigned short* gp1 = gq1 + (size_t)(LEN - 2) * Q * 2;
    for (int t = LEN - 2; t >= 0; --t) {
      f8v z = {0.f, 0.f, 0.f, 0.f, 0.f, 0.f, 0.f, 0.f};
      f8v acc0 = z, acc1 = z;
#pragma unroll
      for (int kt = 0; kt < KC; ++kt) {
        h16v v = ldrowfrag(Vh, kt, lane);
        acc0 = wmma_f16(v, cA0[kt], acc0);
        acc1 = wmma_f16(v, cA1[kt], acc1);
      }
#pragma unroll 4
      for (int kt = KC; kt < KTL; ++kt) {
        h16v v  = ldrowfrag(Vh, kt, lane);
        h16v b0 = ldfrag(Am + ((size_t)ncol[0] * KTL + kt) * 512, lane);
        h16v b1 = ldfrag(Am + ((size_t)ncol[1] * KTL + kt) * 512, lane);
        acc0 = wmma_f16(v, b0, acc0);
        acc1 = wmma_f16(v, b1, acc1);
      }
      f8v em0 = wmma_f16(a, cBm0, z);
      f8v em1 = wmma_f16(a, cBm1, z);
#pragma unroll
      for (int r = 0; r < 8; ++r) {
        float S0 = acc0[r], S1 = acc1[r];
        gp0[(size_t)r * RSTRIDE + 1] = f2bf(S0);
        gp1[(size_t)r * RSTRIDE + 1] = f2bf(S1);
        pb0[r * Q] = S0 * em0[r];
        pb1[r * Q] = S1 * em1[r];
      }
      int tn = (t > 0) ? t - 1 : 0;
      a = ldfrag(inF + (size_t)tn * 1024, lane);
      gp0 -= Q * 2;
      gp1 -= Q * 2;
      __syncthreads();
      {
        float mx = fuse_reduce_scale(Pb, Vh, tid);
        if (lane == 0) rmG[t] = rmacc;
        rmacc += __logf(mx);
      }
      __syncthreads();
    }
  }
}

// ---------- finalize: posterior = rmF + rmB + log(P_f*Sb + eps) - loglik ----------

__global__ void __launch_bounds__(256)
hmm_finalize(float* __restrict__ dOut,
             const float* __restrict__ rmFg,
             const float* __restrict__ rmBg,
             const float* __restrict__ loglikWs) {
  size_t idx = (size_t)blockIdx.x * 256 + threadIdx.x;
  if (idx >= POST_ELEMS) return;
  size_t row = idx >> 9;        // (m*BT + b)*LEN + t
  size_t mb  = row >> 9;        // m*BT + b
  unsigned w = ((const unsigned*)dOut)[idx];
  float pf = __uint_as_float(w << 16);
  float sb = __uint_as_float(w & 0xFFFF0000u);
  float v = pf * sb;
  dOut[idx] = rmFg[row] + rmBg[row] + __logf(v + 1e-37f) - loglikWs[mb];
}

// ---------- launch ----------

extern "C" void kernel_launch(void* const* d_in, const int* in_sizes, int n_in,
                              void* d_out, int out_size, void* d_ws, size_t ws_size,
                              hipStream_t stream) {
  const float* inputs     = (const float*)d_in[0];
  const float* transition = (const float*)d_in[1];
  const float* emission   = (const float*)d_in[2];
  const float* initl      = (const float*)d_in[3];

  char* ws = (char*)d_ws;
  _Float16* Afwd   = (_Float16*)(ws);                                  // 1 MB
  _Float16* Abwd   = (_Float16*)(ws + (1ull << 20));                   // 1 MB
  _Float16* inFrag = (_Float16*)(ws + (2ull << 20));                   // 2 MB
  _Float16* BmFrag = (_Float16*)(ws + (4ull << 20));                   // 64 KB
  float*    pInit  = (float*)   (ws + (4ull << 20) + (64ull << 10));   // 4 KB
  float*    llk    = (float*)   (ws + (4ull << 20) + (68ull << 10));   // 4 KB slot
  float*    rmFg   = (float*)   (ws + (4ull << 20) + (72ull << 10));   // 128 KB
  float*    rmBg   = (float*)   (ws + (4ull << 20) + (200ull << 10));  // 128 KB
  float* out = (float*)d_out;

  prep_trans <<<dim3(2),   dim3(512), 0, stream>>>(transition, Afwd, Abwd);
  prep_emis  <<<dim3(2),   dim3(512), 0, stream>>>(emission, BmFrag);
  prep_inputs<<<dim3(128), dim3(256), 0, stream>>>(inputs, inFrag);
  prep_init  <<<dim3(1),   dim3(32),  0, stream>>>(initl, pInit);

  hmm_chains<<<dim3(8), dim3(512), 0, stream>>>(Afwd, Abwd, inFrag, BmFrag,
                                                pInit, out, llk, rmFg, rmBg);

  unsigned fin_blocks = (unsigned)((POST_ELEMS + 255) / 256);
  hmm_finalize<<<dim3(fin_blocks), dim3(256), 0, stream>>>(out, rmFg, rmBg, llk);
}
